// Attention_26182120637036
// MI455X (gfx1250) — compile-verified, matrix operand order checked by reading, not run on hardware
//
#include <hip/hip_runtime.h>
#include <hip/hip_bf16.h>

typedef __attribute__((ext_vector_type(16))) _Float16 v16h;
typedef __attribute__((ext_vector_type(8)))  float    v8f;

constexpr int BB = 4, NN = 2048, DD = 1024, HH = 16, DH = 64, INNER = 1024;
constexpr int ROWS = BB * NN;            // 8192
constexpr float LN_EPS = 1e-5f;

// Fragment K index for 16-bit WMMA A/B operands (ISA 7.12.2)
static __device__ __forceinline__ int kpat(int i, int half) {
  return ((i >> 3) << 4) + (half << 3) + (i & 7);
}
// Position of k (within a 32-k group) in fragment-permuted storage:
// perm32(kpat(i,half)) == half*16 + i
static __device__ __forceinline__ int perm32(int g) {
  return (((g >> 3) & 1) << 4) | (((g >> 4) & 1) << 3) | (g & 7);
}

static __device__ __forceinline__ v8f wmma_f16(v16h a, v16h b, v8f c) {
  return __builtin_amdgcn_wmma_f32_16x16x32_f16(false, a, false, b, (short)0, c,
                                                false, false);
}

// ------------------------------------------------ LayerNorm(x) -> f16, A-frag permuted rows
__global__ void ln_x_kernel(const float* __restrict__ x, const float* __restrict__ w,
                            const float* __restrict__ b, _Float16* __restrict__ out) {
  const int row  = blockIdx.x * 8 + (threadIdx.x >> 5);
  const int lane = threadIdx.x & 31;
  const float* xr = x + (size_t)row * DD;
  float s = 0.f, s2 = 0.f;
  for (int i = lane; i < DD; i += 32) { float v = xr[i]; s += v; s2 += v * v; }
#pragma unroll
  for (int off = 16; off; off >>= 1) {
    s  += __shfl_xor(s,  off, 32);
    s2 += __shfl_xor(s2, off, 32);
  }
  const float mu   = s * (1.f / DD);
  const float var  = s2 * (1.f / DD) - mu * mu;
  const float rstd = rsqrtf(var + LN_EPS);
  _Float16* orow = out + (size_t)row * DD;
  const int pp = perm32(lane);                  // i & 31 == lane for all iterations
  for (int i = lane; i < DD; i += 32)
    orow[(i & ~31) + pp] = (_Float16)((xr[i] - mu) * rstd * w[i] + b[i]);
}

// ------------------------------------------------ W (1024x1024 f32) -> fragment-order f16
__global__ void cvt_w_frag_kernel(const float* __restrict__ W, _Float16* __restrict__ out) {
  const int t = blockIdx.x * blockDim.x + threadIdx.x;   // 0 .. 256K-1
  const int k = t >> 8;
  const int n0 = (t & 255) * 4;
  const int kc = k >> 5, k5 = k & 31;
  const int hf = (k5 >> 3) & 1;
  const int ii = ((k5 >> 4) << 3) | (k5 & 7);
  const float4 v = *(const float4*)(W + (size_t)k * 1024 + n0);
  const float* vp = (const float*)&v;
#pragma unroll
  for (int e = 0; e < 4; ++e) {
    const int n = n0 + e;
    const int ny = n >> 7, ns = (n >> 4) & 7, l16 = n & 15;
    const int lane = l16 + (hf << 4);
    out[((((size_t)ny * 32 + kc) * 8 + ns) * 32 + lane) * 16 + ii] = (_Float16)vp[e];
  }
}

// ------------------------------------------------ GEMM fragment set + WMMA step helpers
struct GFrag { v16h a0, a1, b0, b1, b2, b3; };

static __device__ __forceinline__ GFrag gload(const _Float16* a0p, const _Float16* a1p,
                                              const _Float16* wb) {
  GFrag f;
  f.a0 = *(const v16h*)(a0p);
  f.a1 = *(const v16h*)(a1p);
  f.b0 = *(const v16h*)(wb);
  f.b1 = *(const v16h*)(wb + 512);
  f.b2 = *(const v16h*)(wb + 1024);
  f.b3 = *(const v16h*)(wb + 1536);
  return f;
}
static __device__ __forceinline__ void gstep(const GFrag& f, v8f acc[2][4]) {
  acc[0][0] = wmma_f16(f.a0, f.b0, acc[0][0]);
  acc[0][1] = wmma_f16(f.a0, f.b1, acc[0][1]);
  acc[0][2] = wmma_f16(f.a0, f.b2, acc[0][2]);
  acc[0][3] = wmma_f16(f.a0, f.b3, acc[0][3]);
  acc[1][0] = wmma_f16(f.a1, f.b0, acc[1][0]);
  acc[1][1] = wmma_f16(f.a1, f.b1, acc[1][1]);
  acc[1][2] = wmma_f16(f.a1, f.b2, acc[1][2]);
  acc[1][3] = wmma_f16(f.a1, f.b3, acc[1][3]);
}

// ------------------------------------------------ QKV projection GEMM (128x128 tile, no LDS)
// Wave owns 32 rows x 64 cols (= one full head slice) -> per-head LN done with
// intra-wave shuffle reductions in the WMMA C-layout, scatter to frag layouts.
// mode 0: q -> Q' (LN*0.125); mode 1: k -> K' (LN); mode 2: v -> V' (plain)
__global__ void gemm_qkv_kernel(const _Float16* __restrict__ Ap,
                                const _Float16* __restrict__ Wf,
                                const float* __restrict__ lnw,
                                const float* __restrict__ lnb,
                                _Float16* __restrict__ out, int mode) {
  const int tid = threadIdx.x;
  const int wv = tid >> 5, lane = tid & 31, lane16 = lane & 15, half = lane >> 4;
  const int rp = wv & 3, nb = wv >> 2;
  const int m0 = blockIdx.x * 128;
  const int h = blockIdx.y * 2 + nb;           // this wave's head

  const _Float16* arow0 = Ap + (size_t)(m0 + rp * 32 + lane16) * DD + half * 16;
  const _Float16* arow1 = arow0 + (size_t)16 * DD;
  const _Float16* wbase =
      Wf + (((size_t)blockIdx.y * 32 * 8) + nb * 4) * 512 + (size_t)lane * 16;

  v8f acc[2][4] = {};

  GFrag fA = gload(arow0, arow1, wbase);
#pragma unroll 1
  for (int kc = 0; kc < 30; kc += 2) {
    GFrag fB = gload(arow0 + (kc + 1) * 32, arow1 + (kc + 1) * 32,
                     wbase + (size_t)(kc + 1) * 4096);
    gstep(fA, acc);
    fA = gload(arow0 + (kc + 2) * 32, arow1 + (kc + 2) * 32,
               wbase + (size_t)(kc + 2) * 4096);
    gstep(fB, acc);
  }
  {
    GFrag fB = gload(arow0 + 31 * 32, arow1 + 31 * 32, wbase + (size_t)31 * 4096);
    gstep(fA, acc);
    gstep(fB, acc);
  }

  // -------- epilogue: in-register per-head LN + scatter to fragment layouts
  const float scl = (mode == 0) ? 0.125f : 1.0f;
  float lw[4], lb[4];
#pragma unroll
  for (int nt = 0; nt < 4; ++nt) {
    lw[nt] = lnw[nt * 16 + lane16];
    lb[nt] = lnb[nt * 16 + lane16];
  }

#pragma unroll
  for (int rs = 0; rs < 2; ++rs)
#pragma unroll
    for (int r = 0; r < 8; ++r) {
      const int mg = m0 + rp * 32 + rs * 16 + r + half * 8;
      const int b = mg >> 11, nrow = mg & (NN - 1);
      const int bh = b * HH + h;
      float vals[4];
      if (mode != 2) {
        float s = 0.f, s2 = 0.f;
#pragma unroll
        for (int nt = 0; nt < 4; ++nt) {
          const float v = acc[rs][nt][r];
          s += v; s2 += v * v;
        }
#pragma unroll
        for (int off = 1; off < 16; off <<= 1) {
          s  += __shfl_xor(s,  off, 32);
          s2 += __shfl_xor(s2, off, 32);
        }
        const float mu = s * (1.f / 64.f);
        const float rstd = rsqrtf(s2 * (1.f / 64.f) - mu * mu + LN_EPS);
#pragma unroll
        for (int nt = 0; nt < 4; ++nt)
          vals[nt] = ((acc[rs][nt][r] - mu) * rstd * lw[nt] + lb[nt]) * scl;
      } else {
#pragma unroll
        for (int nt = 0; nt < 4; ++nt) vals[nt] = acc[rs][nt][r];
      }

      if (mode == 0) {                         // Q' frag layout
        const int qb = nrow >> 6, wq = (nrow >> 4) & 3, l16r = nrow & 15;
        _Float16* base = out + ((((size_t)bh * 32 + qb) * 4 + wq) * 32) * 32;
#pragma unroll
        for (int nt = 0; nt < 4; ++nt) {
          const int dh = nt * 16 + lane16;
          const int c = dh >> 5, d5 = dh & 31;
          const int hf = (d5 >> 3) & 1, ii = ((d5 >> 4) << 3) | (d5 & 7);
          base[(size_t)(l16r + (hf << 4)) * 32 + c * 16 + ii] = (_Float16)vals[nt];
        }
      } else if (mode == 1) {                  // K' frag layout
        const int jb = nrow >> 6, ntk = (nrow >> 4) & 3, l16r = nrow & 15;
        _Float16* base = out + (((size_t)bh * 32 + jb) * 8) * 512;
#pragma unroll
        for (int nt = 0; nt < 4; ++nt) {
          const int dh = nt * 16 + lane16;
          const int c = dh >> 5, d5 = dh & 31;
          const int hf = (d5 >> 3) & 1, ii = ((d5 >> 4) << 3) | (d5 & 7);
          base[(size_t)(c * 4 + ntk) * 512 + (l16r + (hf << 4)) * 16 + ii] =
              (_Float16)vals[nt];
        }
      } else {                                 // V' frag layout
        const int jb = nrow >> 6, kl = nrow & 63;
        const int cv = kl >> 5, k5 = kl & 31;
        const int hfv = (k5 >> 3) & 1, iiv = ((k5 >> 4) << 3) | (k5 & 7);
        _Float16* base = out + (((size_t)bh * 32 + jb) * 8) * 512;
#pragma unroll
        for (int nt = 0; nt < 4; ++nt)
          base[(size_t)(cv * 4 + nt) * 512 + (lane16 + (hfv << 4)) * 16 + iiv] =
              (_Float16)vals[nt];
      }
    }
}

// ------------------------------------------------ causal flash attention (frag-layout I/O)
__global__ void flash_attn_kernel(const _Float16* __restrict__ Qf,
                                  const _Float16* __restrict__ Kf,
                                  const _Float16* __restrict__ Vf,
                                  _Float16* __restrict__ Op) {
  __shared__ _Float16 sP[4][16 * 80];          // per-wave P tile, perm32 row layout

  const int tid = threadIdx.x;
  const int wv = tid >> 5, lane = tid & 31, lane16 = lane & 15, half = lane >> 4;
  const int qblk = blockIdx.x;
  const int bh = blockIdx.y;
  const int b = bh >> 4, h = bh & 15;

  const _Float16* qp = Qf + ((((size_t)bh * 32 + qblk) * 4 + wv) * 32 + lane) * 32;
  const v16h aq0 = *(const v16h*)(qp);
  const v16h aq1 = *(const v16h*)(qp + 16);

  float mrow[8], lrow[8];
  v8f oacc[4] = {};
#pragma unroll
  for (int r = 0; r < 8; ++r) { mrow[r] = -1e30f; lrow[r] = 0.f; }

  _Float16* sPw = sP[wv];
  int pcol[4];
#pragma unroll
  for (int nt = 0; nt < 4; ++nt)
    pcol[nt] = ((nt >> 1) << 5) + perm32(((nt & 1) << 4) | lane16);

  for (int j = 0; j <= qblk; ++j) {
    const _Float16* kb = Kf + ((size_t)(bh * 32 + j) * 8) * 512 + (size_t)lane * 16;
    const _Float16* vb = Vf + ((size_t)(bh * 32 + j) * 8) * 512 + (size_t)lane * 16;

    // load all fragments for this tile up front: K first (S waits only on these),
    // V second (covered by softmax latency)
    v16h kfr[8], vfr[8];
#pragma unroll
    for (int g = 0; g < 8; ++g) kfr[g] = *(const v16h*)(kb + g * 512);
#pragma unroll
    for (int g = 0; g < 8; ++g) vfr[g] = *(const v16h*)(vb + g * 512);
    if (j < qblk) {                             // prefetch next tiles
      __builtin_prefetch(kb + 8 * 512, 0, 0);
      __builtin_prefetch(vb + 8 * 512, 0, 0);
    }

    v8f sacc[4] = {};
#pragma unroll
    for (int c = 0; c < 2; ++c) {
      const v16h a = c ? aq1 : aq0;
#pragma unroll
      for (int nt = 0; nt < 4; ++nt)
        sacc[nt] = wmma_f16(a, kfr[c * 4 + nt], sacc[nt]);
    }

    if (j == qblk) {                            // causal mask on diagonal block
#pragma unroll
      for (int nt = 0; nt < 4; ++nt)
#pragma unroll
        for (int r = 0; r < 8; ++r) {
          const int kj = j * 64 + nt * 16 + lane16;
          const int qi = qblk * 64 + wv * 16 + r + half * 8;
          if (kj > qi) sacc[nt][r] = -1e30f;
        }
    }

#pragma unroll
    for (int r = 0; r < 8; ++r) {
      float pm = fmaxf(fmaxf(sacc[0][r], sacc[1][r]), fmaxf(sacc[2][r], sacc[3][r]));
#pragma unroll
      for (int off = 1; off < 16; off <<= 1) pm = fmaxf(pm, __shfl_xor(pm, off, 32));
      const float mn = fmaxf(mrow[r], pm);
      const float al = __expf(mrow[r] - mn);
      float rs = 0.f;
      const int prow = (r + half * 8) * 80;
#pragma unroll
      for (int nt = 0; nt < 4; ++nt) {
        const float p = __expf(sacc[nt][r] - mn);
        rs += p;
        sPw[prow + pcol[nt]] = (_Float16)p;
        oacc[nt][r] *= al;
      }
#pragma unroll
      for (int off = 1; off < 16; off <<= 1) rs += __shfl_xor(rs, off, 32);
      lrow[r] = lrow[r] * al + rs;
      mrow[r] = mn;
    }

#pragma unroll
    for (int c = 0; c < 2; ++c) {               // O += P @ V
      const v16h ap = *(const v16h*)(sPw + lane16 * 80 + c * 32 + half * 16);
#pragma unroll
      for (int nt = 0; nt < 4; ++nt)
        oacc[nt] = wmma_f16(ap, vfr[c * 4 + nt], oacc[nt]);
    }
  }

  // epilogue: write in A-frag permuted row order for the output projection
#pragma unroll
  for (int nt = 0; nt < 4; ++nt)
#pragma unroll
    for (int r = 0; r < 8; ++r) {
      const int nrow = qblk * 64 + wv * 16 + r + half * 8;
      const int col = h * 64 + nt * 16 + lane16;
      const int pos = (col & ~31) + perm32(col & 31);
      Op[((size_t)b * NN + nrow) * INNER + pos] = (_Float16)(oacc[nt][r] / lrow[r]);
    }
}

// ------------------------------------------------ output projection (128x128, no LDS)
__global__ void gemm_o_kernel(const _Float16* __restrict__ Ap,
                              const _Float16* __restrict__ Wf,
                              const float* __restrict__ bias,
                              float* __restrict__ out) {
  const int tid = threadIdx.x;
  const int wv = tid >> 5, lane = tid & 31, lane16 = lane & 15, half = lane >> 4;
  const int rp = wv & 3, nb = wv >> 2;
  const int m0 = blockIdx.x * 128;

  const _Float16* arow0 = Ap + (size_t)(m0 + rp * 32 + lane16) * INNER + half * 16;
  const _Float16* arow1 = arow0 + (size_t)16 * INNER;
  const _Float16* wbase =
      Wf + (((size_t)blockIdx.y * 32 * 8) + nb * 4) * 512 + (size_t)lane * 16;

  v8f acc[2][4] = {};

  GFrag fA = gload(arow0, arow1, wbase);
#pragma unroll 1
  for (int kc = 0; kc < 30; kc += 2) {
    GFrag fB = gload(arow0 + (kc + 1) * 32, arow1 + (kc + 1) * 32,
                     wbase + (size_t)(kc + 1) * 4096);
    gstep(fA, acc);
    fA = gload(arow0 + (kc + 2) * 32, arow1 + (kc + 2) * 32,
               wbase + (size_t)(kc + 2) * 4096);
    gstep(fB, acc);
  }
  {
    GFrag fB = gload(arow0 + 31 * 32, arow1 + 31 * 32, wbase + (size_t)31 * 4096);
    gstep(fA, acc);
    gstep(fB, acc);
  }

  float bs[4];
#pragma unroll
  for (int nt = 0; nt < 4; ++nt)
    bs[nt] = bias[blockIdx.y * 128 + nb * 64 + nt * 16 + lane16];

#pragma unroll
  for (int rs = 0; rs < 2; ++rs)
#pragma unroll
    for (int nt = 0; nt < 4; ++nt) {
      const int ng = blockIdx.y * 128 + nb * 64 + nt * 16 + lane16;
#pragma unroll
      for (int r = 0; r < 8; ++r) {
        const int mg = m0 + rp * 32 + rs * 16 + r + half * 8;
        out[(size_t)mg * DD + ng] = acc[rs][nt][r] + bs[nt];
      }
    }
}

// ------------------------------------------------ host launcher
extern "C" void kernel_launch(void* const* d_in, const int* in_sizes, int n_in,
                              void* d_out, int out_size, void* d_ws, size_t ws_size,
                              hipStream_t stream) {
  (void)in_sizes; (void)n_in; (void)out_size; (void)ws_size;
  const float* x    = (const float*)d_in[0];
  const float* ln_w = (const float*)d_in[1];
  const float* ln_b = (const float*)d_in[2];
  const float* Wq   = (const float*)d_in[3];
  const float* Wk   = (const float*)d_in[4];
  const float* Wv   = (const float*)d_in[5];
  const float* qn_w = (const float*)d_in[6];
  const float* qn_b = (const float*)d_in[7];
  const float* kn_w = (const float*)d_in[8];
  const float* kn_b = (const float*)d_in[9];
  const float* Wo   = (const float*)d_in[10];
  const float* bo   = (const float*)d_in[11];

  char* ws = (char*)d_ws;
  const size_t MB = 1024 * 1024;
  _Float16* xnp = (_Float16*)(ws);                   // 16 MB, permuted rows
  _Float16* wqf = (_Float16*)(ws + 16 * MB);         // 2 MB each, fragment order
  _Float16* wkf = wqf + (size_t)DD * INNER;
  _Float16* wvf = wkf + (size_t)DD * INNER;
  _Float16* wof = wvf + (size_t)DD * INNER;
  _Float16* qf  = (_Float16*)(ws + 24 * MB);         // 16 MB each, fragment order
  _Float16* kf  = qf + (size_t)ROWS * INNER;
  _Float16* vf  = kf + (size_t)ROWS * INNER;
  _Float16* atp = vf + (size_t)ROWS * INNER;         // permuted rows

  ln_x_kernel<<<ROWS / 8, 256, 0, stream>>>(x, ln_w, ln_b, xnp);

  cvt_w_frag_kernel<<<1024, 256, 0, stream>>>(Wq, wqf);
  cvt_w_frag_kernel<<<1024, 256, 0, stream>>>(Wk, wkf);
  cvt_w_frag_kernel<<<1024, 256, 0, stream>>>(Wv, wvf);
  cvt_w_frag_kernel<<<1024, 256, 0, stream>>>(Wo, wof);

  dim3 ggrid(ROWS / 128, INNER / 128);
  gemm_qkv_kernel<<<ggrid, 256, 0, stream>>>(xnp, wqf, qn_w, qn_b, qf, 0);
  gemm_qkv_kernel<<<ggrid, 256, 0, stream>>>(xnp, wkf, kn_w, kn_b, kf, 1);
  gemm_qkv_kernel<<<ggrid, 256, 0, stream>>>(xnp, wvf, qn_w, qn_b, vf, 2);

  flash_attn_kernel<<<dim3(NN / 64, BB * HH), 128, 0, stream>>>(qf, kf, vf, atp);

  gemm_o_kernel<<<dim3(ROWS / 128, DD / 128), 256, 0, stream>>>(atp, wof, bo,
                                                                (float*)d_out);
}
